// UnrolledFBS_86079734547169
// MI455X (gfx1250) — compile-verified, matrix operand order checked by reading, not run on hardware
//
#include <hip/hip_runtime.h>
#include <math.h>

// ---------------------------------------------------------------------------
// UnrolledFBS on MI455X (gfx1250, wave32, WMMA + TDM).
//
// Per step: elementwise prox update (+ res/delta block partials)  ->
//   GEMM1 bf16-WMMA [16384,2304]x[2304,256] relu ->
//   GEMM2 [16384,256]x[256,256] relu ->
//   GEMM3 [16384,256]x[256,1152] f32 out (+ Q block partials) ->
//   1-block finalize (res/delta/Q sums -> alpha scalar, residual[n]).
//
// GEMM data flow: each block's 16xK bf16 activation slab is DMA'd into LDS
// once by the Tensor Data Mover (tensor_load_to_lds + s_wait_tensorcnt),
// shared by all 4 waves (A fragments via ds_load_b128). Weights (L2-resident)
// stream from global in fragment-major layout with a branch-free ping-pong
// K-loop (Ktiles is even for all three layers; the one overshooting load per
// tile hits a padded region of the workspace and is never consumed).
// ---------------------------------------------------------------------------

typedef __attribute__((ext_vector_type(16))) __bf16 v16bf;
typedef __attribute__((ext_vector_type(8)))  float  v8f;
typedef __attribute__((ext_vector_type(4)))  unsigned int u32x4;
typedef __attribute__((ext_vector_type(4)))  unsigned int tdm_u32x4;
typedef __attribute__((ext_vector_type(8)))  int tdm_i32x8;
typedef __attribute__((ext_vector_type(4)))  int tdm_i32x4;

#define B_ROWS   16384
#define S_ELEMS  576            // N_CH * 8 * 8
#define IN_F     2304
#define HIDDEN   256
#define OUT_F    1152
#define STATE_N  ((size_t)B_ROWS * S_ELEMS)   // 9437184
#define T_STEPS  20
#define TAU_C    0.1f
#define ZETA_C   0.99f

union FragU { u32x4 q[2]; v16bf v; };

__device__ __forceinline__ unsigned short f2bf(float x) {
  unsigned int u = __float_as_uint(x);
  u += 0x7FFFu + ((u >> 16) & 1u);          // round-to-nearest-even
  return (unsigned short)(u >> 16);
}

// ---------------------------------------------------------------------------
// TDM: 1-D contiguous copy global -> LDS. n8 = size in 8-byte units (<=65535).
// ---------------------------------------------------------------------------
__device__ __forceinline__ void tdm_copy_to_lds_1d(const void* gsrc,
                                                   unsigned lds_off, unsigned n8) {
  unsigned long long ga = (unsigned long long)(uintptr_t)gsrc;
  tdm_u32x4 g0;
  g0[0] = 1u;                                              // count=1 (user D#)
  g0[1] = lds_off;                                         // lds_addr (bytes)
  g0[2] = (unsigned)ga;                                    // global_addr[31:0]
  g0[3] = (unsigned)((ga >> 32) & 0x01FFFFFFu) | (2u << 30); // addr[56:32]|type=2
  tdm_i32x8 g1;
  g1[0] = (int)(3u << 16);                                 // data_size = 8 bytes
  g1[1] = (int)((n8 & 0xFFFFu) << 16);                     // tensor_dim0[15:0]
  g1[2] = (int)(((n8 >> 16) & 0xFFFFu) | (1u << 16));      // td0[31:16]|tensor_dim1=1
  g1[3] = (int)((n8 & 0xFFFFu) << 16);                     // tile_dim0
  g1[4] = 1;                                               // tile_dim1=1, tile_dim2=0
  g1[5] = (int)n8;                                         // tensor_dim0_stride lo32
  g1[6] = 0;
  g1[7] = 0;
  tdm_i32x4 z4 = {};
#if __has_builtin(__builtin_amdgcn_tensor_load_to_lds)
#if defined(__clang_major__) && (__clang_major__ >= 23)
  tdm_i32x8 z8 = {};
  __builtin_amdgcn_tensor_load_to_lds(g0, g1, z4, z4, z8, 0);
#else
  __builtin_amdgcn_tensor_load_to_lds(g0, g1, z4, z4, 0);
#endif
  __builtin_amdgcn_s_wait_tensorcnt(0);
#else
  (void)g0; (void)g1; (void)z4;
#endif
}

// Stage one block's A-slab (bytes, 16-byte multiple) into dynamic LDS.
__device__ __forceinline__ void stage_a_slab(char* smem, const void* gsrc,
                                             unsigned bytes) {
#if __has_builtin(__builtin_amdgcn_tensor_load_to_lds)
  if (threadIdx.x < 32)    // wave 0 issues the DMA and waits on TENSORcnt
    tdm_copy_to_lds_1d(gsrc, __builtin_amdgcn_groupstaticsize(), bytes >> 3);
  __syncthreads();
#else
  const u32x4* g = (const u32x4*)gsrc;
  u32x4* s = (u32x4*)smem;
  for (unsigned i = threadIdx.x; i < (bytes >> 4); i += blockDim.x) s[i] = g[i];
  __syncthreads();
#endif
}

// A fragment (16x32 bf16) from LDS slab. Lane L: m=L&15, hi=L>>4;
// k = kbase + hi*8 + 0..7 and kbase + 16 + hi*8 + 0..7 => 2x ds_load_b128.
__device__ __forceinline__ v16bf ldfrag_a_lds(const char* smem, unsigned K,
                                              int lane, unsigned kbase) {
  unsigned m = (unsigned)(lane & 15), hi = (unsigned)(lane >> 4);
  const char* p = smem + (size_t)((m * K + hi * 8u + kbase) * 2u);
  FragU f;
  f.q[0] = *(const u32x4*)(p);
  f.q[1] = *(const u32x4*)(p + 32);
  return f.v;
}

// B fragment from fragment-major packed weights (uniform base + 32-bit offset).
__device__ __forceinline__ FragU ldfrag_b(const unsigned short* __restrict__ Wp,
                                          unsigned off_elems) {
  FragU f;
  f.q[0] = *(const u32x4*)(Wp + off_elems);
  f.q[1] = *(const u32x4*)(Wp + off_elems + 8);
  return f;
}

#define WMMA_BF16(ACC, A, B) \
  __builtin_amdgcn_wmma_f32_16x16x32_bf16(false, (A), false, (B), (short)0, (ACC), false, false)

// ---------------------------------------------------------------------------
// Weight packing: row-major f32 W[K,N] -> B-fragment-major bf16.
// ---------------------------------------------------------------------------
__global__ void pack_w_kernel(const float* __restrict__ W,
                              unsigned short* __restrict__ Wp, int K, int N) {
  size_t e = (size_t)blockIdx.x * 256 + threadIdx.x;
  if (e >= (size_t)K * N) return;
  int Ntiles = N >> 4;
  int frag = (int)(e >> 9);
  int within = (int)(e & 511);
  int lane = within >> 4;
  int t = within & 15;
  int j = t >> 1, bbit = t & 1;
  int kt = frag / Ntiles, nt = frag % Ntiles;
  int n = nt * 16 + (lane & 15);
  int hi = lane >> 4;
  int k = kt * 32 + ((j & 4) ? 16 : 0) + hi * 8 + (j & 3) * 2 + bbit;
  Wp[e] = f2bf(W[(size_t)k * N + n]);
}

__global__ void init_state_kernel(const float* __restrict__ noisy,
                                  float* __restrict__ x, float* __restrict__ y,
                                  float* __restrict__ p, float* __restrict__ z,
                                  float* __restrict__ scalars) {
  size_t i = (size_t)blockIdx.x * 256 + threadIdx.x;
  if (i >= STATE_N) return;
  size_t row = i / S_ELEMS;
  int col = (int)(i - row * S_ELEMS);
  float v = (col < 64) ? noisy[row * 64 + col] : 0.0f;
  x[i] = v; y[i] = v; p[i] = v; z[i] = v;
  if (i == 0) scalars[0] = 0.0f;   // alpha=0 -> u=v=0 at step 0
}

// ---------------------------------------------------------------------------
// Elementwise step: y,z,pr,x_new + res/delta partials + bf16 MLP input.
// gamma = lambda = 1, a_n == ab_n = n/(n+3).
// ---------------------------------------------------------------------------
__global__ __launch_bounds__(256)
void step_elem_kernel(float* __restrict__ x, float* __restrict__ y,
                      float* __restrict__ p, float* __restrict__ z,
                      const float* __restrict__ uv,      // [B,1152] u_raw||v_raw
                      const float* __restrict__ scalars, // [0] = alpha
                      unsigned short* __restrict__ inp,  // [B,2304] bf16
                      float a_n,
                      float* __restrict__ resP, float* __restrict__ deltaP,
                      float* __restrict__ p_out) {       // d_out on last step
  __shared__ float sres[256], sdel[256];
  const float alpha = scalars[0];
  float lres = 0.0f, ldel = 0.0f;
  int tid = threadIdx.x;
  size_t base = (size_t)blockIdx.x * 1024;
  #pragma unroll
  for (int kk = 0; kk < 4; ++kk) {
    size_t i = base + (size_t)kk * 256 + tid;
    size_t row = i / S_ELEMS;
    int col = (int)(i - row * S_ELEMS);
    float xv = x[i], yp = y[i], pp = p[i], zp = z[i];
    float u = alpha * uv[row * 1152 + col];
    float vv = alpha * uv[row * 1152 + 576 + col];
    float yn = xv + a_n * (yp - xv) + u;
    float zn = xv + a_n * (pp - xv) + a_n * (zp - pp) + u + vv;
    float zm = zn - yn;                                   // z - gam*y
    float pr = copysignf(fmaxf(fabsf(zm) - TAU_C, 0.0f), zm);
    float xn = xv + (pr - zn) + a_n * (zp - pp);
    float dr = pr - yn;  lres += dr * dr;
    float dd = xn - zn;  ldel += dd * dd;
    x[i] = xn; y[i] = yn; p[i] = pr; z[i] = zn;
    size_t ib = row * (size_t)IN_F;
    inp[ib + col]         = f2bf(xn);
    inp[ib + 576 + col]   = f2bf(pr);
    inp[ib + 1152 + col]  = f2bf(yn);
    inp[ib + 1728 + col]  = f2bf(zn);
    if (p_out) p_out[i] = pr;
  }
  sres[tid] = lres; sdel[tid] = ldel;
  __syncthreads();
  for (int s = 128; s > 0; s >>= 1) {
    if (tid < s) { sres[tid] += sres[tid + s]; sdel[tid] += sdel[tid + s]; }
    __syncthreads();
  }
  if (tid == 0) { resP[blockIdx.x] = sres[0]; deltaP[blockIdx.x] = sdel[0]; }
}

// ---------------------------------------------------------------------------
// bf16 WMMA GEMM: Out_bf16[M,N] = relu?(A[M,K] @ Wp + bias).
// Block = 4 waves; wave w -> 4 N-tiles (16x64 strip); grid.x = M/16.
// Branch-free ping-pong K-loop (Ktiles even); A from LDS, B from global.
// ---------------------------------------------------------------------------
template<bool RELU>
__global__ __launch_bounds__(128)
void gemm_bf16_kernel(const unsigned short* __restrict__ A,
                      const unsigned short* __restrict__ Wp,
                      const float* __restrict__ bias,
                      unsigned short* __restrict__ Out,
                      int K, int N) {
  extern __shared__ char smem[];
  const int lane = threadIdx.x & 31;
  const int wave = threadIdx.x >> 5;
  const int rowbase = blockIdx.x * 16;
  const unsigned Ktiles = (unsigned)K >> 5;   // even (72 or 8)
  const unsigned Ntiles = (unsigned)N >> 4;

  stage_a_slab(smem, A + (size_t)rowbase * K, (unsigned)(16 * K * 2));

  const unsigned bstride = Ntiles * 512u;     // elems per K-chunk
  unsigned boff = (unsigned)lane * 16u + (unsigned)(wave * 4) * 512u;
  FragU ba0 = ldfrag_b(Wp, boff);
  FragU ba1 = ldfrag_b(Wp, boff + 512u);
  FragU ba2 = ldfrag_b(Wp, boff + 1024u);
  FragU ba3 = ldfrag_b(Wp, boff + 1536u);
  FragU bb0, bb1, bb2, bb3;
  v8f acc0 = {}, acc1 = {}, acc2 = {}, acc3 = {};
  for (unsigned kt = 0; kt < Ktiles; kt += 2) {
    v16bf a0 = ldfrag_a_lds(smem, (unsigned)K, lane, kt * 32u);
    v16bf a1 = ldfrag_a_lds(smem, (unsigned)K, lane, kt * 32u + 32u);
    const unsigned o1 = boff + bstride;       // kt+1 (always in-range)
    bb0 = ldfrag_b(Wp, o1);
    bb1 = ldfrag_b(Wp, o1 + 512u);
    bb2 = ldfrag_b(Wp, o1 + 1024u);
    bb3 = ldfrag_b(Wp, o1 + 1536u);
    acc0 = WMMA_BF16(acc0, a0, ba0.v);
    acc1 = WMMA_BF16(acc1, a0, ba1.v);
    acc2 = WMMA_BF16(acc2, a0, ba2.v);
    acc3 = WMMA_BF16(acc3, a0, ba3.v);
    const unsigned o2 = boff + 2u * bstride;  // kt+2 (pad region on last iter)
    ba0 = ldfrag_b(Wp, o2);
    ba1 = ldfrag_b(Wp, o2 + 512u);
    ba2 = ldfrag_b(Wp, o2 + 1024u);
    ba3 = ldfrag_b(Wp, o2 + 1536u);
    acc0 = WMMA_BF16(acc0, a1, bb0.v);
    acc1 = WMMA_BF16(acc1, a1, bb1.v);
    acc2 = WMMA_BF16(acc2, a1, bb2.v);
    acc3 = WMMA_BF16(acc3, a1, bb3.v);
    boff += 2u * bstride;
  }
  // D layout: lane -> col=lane&15, half=lane>>4; reg r -> m = r + 8*half.
  const int half = lane >> 4, col = lane & 15;
  unsigned short* obase = Out + (size_t)rowbase * N;
  #pragma unroll
  for (int t = 0; t < 4; ++t) {
    v8f acc = (t == 0) ? acc0 : (t == 1) ? acc1 : (t == 2) ? acc2 : acc3;
    int n = (wave * 4 + t) * 16 + col;
    float bv = bias[n];
    #pragma unroll
    for (int r = 0; r < 8; ++r) {
      float d = acc[r] + bv;
      if (RELU) d = fmaxf(d, 0.0f);
      obase[(unsigned)(r + half * 8) * (unsigned)N + (unsigned)n] = f2bf(d);
    }
  }
}

// ---------------------------------------------------------------------------
// Final GEMM: f32 out [B,1152] = A[B,256] @ W3 + b3, plus Q partial sums.
// grid = (M/16, 5); block covers 16 N-tiles (72 total, tail clamped).
// ---------------------------------------------------------------------------
__global__ __launch_bounds__(128)
void gemm3_kernel(const unsigned short* __restrict__ A,
                  const unsigned short* __restrict__ Wp,
                  const float* __restrict__ bias,
                  float* __restrict__ Out,
                  float* __restrict__ qPartials,
                  int K, int N) {
  __shared__ float sq[128];
  extern __shared__ char smem[];
  const int lane = threadIdx.x & 31;
  const int wave = threadIdx.x >> 5;
  const int rowbase = blockIdx.x * 16;
  const unsigned Ktiles = (unsigned)K >> 5;   // even (8)
  const unsigned Ntiles = (unsigned)N >> 4;

  stage_a_slab(smem, A + (size_t)rowbase * K, (unsigned)(16 * K * 2));

  int ntb = blockIdx.y * 16 + wave * 4;
  int nt[4]; bool live[4];
  #pragma unroll
  for (int t = 0; t < 4; ++t) {
    int v = ntb + t;
    live[t] = (v < (int)Ntiles);
    nt[t] = live[t] ? v : ((int)Ntiles - 1);  // clamp (wave-uniform, EXEC full)
  }
  const unsigned bstride = Ntiles * 512u;
  unsigned bo0 = (unsigned)lane * 16u + (unsigned)nt[0] * 512u;
  unsigned bo1 = (unsigned)lane * 16u + (unsigned)nt[1] * 512u;
  unsigned bo2 = (unsigned)lane * 16u + (unsigned)nt[2] * 512u;
  unsigned bo3 = (unsigned)lane * 16u + (unsigned)nt[3] * 512u;
  FragU ba0 = ldfrag_b(Wp, bo0);
  FragU ba1 = ldfrag_b(Wp, bo1);
  FragU ba2 = ldfrag_b(Wp, bo2);
  FragU ba3 = ldfrag_b(Wp, bo3);
  FragU bb0, bb1, bb2, bb3;
  v8f acc0 = {}, acc1 = {}, acc2 = {}, acc3 = {};
  for (unsigned kt = 0; kt < Ktiles; kt += 2) {
    v16bf a0 = ldfrag_a_lds(smem, (unsigned)K, lane, kt * 32u);
    v16bf a1 = ldfrag_a_lds(smem, (unsigned)K, lane, kt * 32u + 32u);
    bb0 = ldfrag_b(Wp, bo0 + bstride);
    bb1 = ldfrag_b(Wp, bo1 + bstride);
    bb2 = ldfrag_b(Wp, bo2 + bstride);
    bb3 = ldfrag_b(Wp, bo3 + bstride);
    acc0 = WMMA_BF16(acc0, a0, ba0.v);
    acc1 = WMMA_BF16(acc1, a0, ba1.v);
    acc2 = WMMA_BF16(acc2, a0, ba2.v);
    acc3 = WMMA_BF16(acc3, a0, ba3.v);
    ba0 = ldfrag_b(Wp, bo0 + 2u * bstride);   // pad region on last iter
    ba1 = ldfrag_b(Wp, bo1 + 2u * bstride);
    ba2 = ldfrag_b(Wp, bo2 + 2u * bstride);
    ba3 = ldfrag_b(Wp, bo3 + 2u * bstride);
    acc0 = WMMA_BF16(acc0, a1, bb0.v);
    acc1 = WMMA_BF16(acc1, a1, bb1.v);
    acc2 = WMMA_BF16(acc2, a1, bb2.v);
    acc3 = WMMA_BF16(acc3, a1, bb3.v);
    bo0 += 2u * bstride; bo1 += 2u * bstride;
    bo2 += 2u * bstride; bo3 += 2u * bstride;
  }
  const int half = lane >> 4, col = lane & 15;
  float* obase = Out + (size_t)rowbase * N;
  float qsum = 0.0f;
  #pragma unroll
  for (int t = 0; t < 4; ++t) {
    if (!live[t]) continue;
    v8f acc = (t == 0) ? acc0 : (t == 1) ? acc1 : (t == 2) ? acc2 : acc3;
    int n = nt[t] * 16 + col;
    float bv = bias[n];
    #pragma unroll
    for (int r = 0; r < 8; ++r) {
      float d = acc[r] + bv;
      obase[(unsigned)(r + half * 8) * (unsigned)N + (unsigned)n] = d;
      qsum += d * d;
    }
  }
  sq[threadIdx.x] = qsum;
  __syncthreads();
  for (int s = 64; s > 0; s >>= 1) {
    if (threadIdx.x < s) sq[threadIdx.x] += sq[threadIdx.x + s];
    __syncthreads();
  }
  if (threadIdx.x == 0)
    qPartials[(size_t)blockIdx.y * gridDim.x + blockIdx.x] = sq[0];
}

// ---------------------------------------------------------------------------
// Per-step scalar finalize: residual[n], alpha for next step.
// ---------------------------------------------------------------------------
__global__ __launch_bounds__(256)
void finalize_kernel(const float* __restrict__ resP, int nres,
                     const float* __restrict__ deltaP, int ndel,
                     const float* __restrict__ qP, int nq,
                     float* __restrict__ scalars,
                     float* __restrict__ res_out) {
  __shared__ float s[256];
  int tid = threadIdx.x;
  float a = 0.0f; for (int i = tid; i < nres; i += 256) a += resP[i];
  s[tid] = a; __syncthreads();
  for (int st = 128; st > 0; st >>= 1) { if (tid < st) s[tid] += s[tid + st]; __syncthreads(); }
  float ressum = s[0]; __syncthreads();
  float b = 0.0f; for (int i = tid; i < ndel; i += 256) b += deltaP[i];
  s[tid] = b; __syncthreads();
  for (int st = 128; st > 0; st >>= 1) { if (tid < st) s[tid] += s[tid + st]; __syncthreads(); }
  float delta = s[0]; __syncthreads();
  float c = 0.0f; for (int i = tid; i < nq; i += 256) c += qP[i];
  s[tid] = c; __syncthreads();
  for (int st = 128; st > 0; st >>= 1) { if (tid < st) s[tid] += s[tid + st]; __syncthreads(); }
  if (tid == 0) {
    float qs = s[0];
    *res_out = sqrtf(ressum + 1e-12f);
    float Q = 1.5f * qs + 1e-12f;
    float budget = ZETA_C * fmaxf(delta, 0.0f);
    scalars[0] = sqrtf(fminf(budget / Q, 1.0f));
  }
}

// ---------------------------------------------------------------------------
extern "C" void kernel_launch(void* const* d_in, const int* in_sizes, int n_in,
                              void* d_out, int out_size, void* d_ws, size_t ws_size,
                              hipStream_t stream) {
  const float* noisy = (const float*)d_in[0];
  const float* W1 = (const float*)d_in[1];
  const float* b1 = (const float*)d_in[2];
  const float* W2 = (const float*)d_in[3];
  const float* b2 = (const float*)d_in[4];
  const float* W3 = (const float*)d_in[5];
  const float* b3 = (const float*)d_in[6];
  float* out = (float*)d_out;                 // [0,9437184) = p_final, then T residuals

  // workspace carve-out (weights padded by one K-chunk stride for the
  // ping-pong loop's overshooting prefetch)
  char* w = (char*)d_ws;
  auto carve = [&](size_t bytes) { void* r = (void*)w; w += (bytes + 255) & ~(size_t)255; return r; };
  float* x  = (float*)carve(STATE_N * 4);
  float* y  = (float*)carve(STATE_N * 4);
  float* p  = (float*)carve(STATE_N * 4);
  float* z  = (float*)carve(STATE_N * 4);
  float* uv = (float*)carve((size_t)B_ROWS * OUT_F * 4);
  unsigned short* inp = (unsigned short*)carve((size_t)B_ROWS * IN_F * 2);
  unsigned short* h1  = (unsigned short*)carve((size_t)B_ROWS * HIDDEN * 2);
  unsigned short* h2  = (unsigned short*)carve((size_t)B_ROWS * HIDDEN * 2);
  const size_t pad16 = (size_t)(HIDDEN / 16) * 512;   // 8192-elem pad (Ntiles=16)
  const size_t pad72 = (size_t)(OUT_F / 16) * 512;    // 36864-elem pad (Ntiles=72)
  unsigned short* W1p = (unsigned short*)carve(((size_t)IN_F * HIDDEN + pad16) * 2);
  unsigned short* W2p = (unsigned short*)carve(((size_t)HIDDEN * HIDDEN + pad16) * 2);
  unsigned short* W3p = (unsigned short*)carve(((size_t)HIDDEN * OUT_F + pad72) * 2);
  float* scalars = (float*)carve(16 * 4);
  const int NB1 = 9216;                       // step_elem blocks (1024 elems each)
  const int NBQ = 1024 * 5;                   // gemm3 blocks
  float* resP   = (float*)carve(NB1 * 4);
  float* deltaP = (float*)carve(NB1 * 4);
  float* qP     = (float*)carve(NBQ * 4);

  // once-per-launch prep (deterministic, re-done every call)
  pack_w_kernel<<<(IN_F * HIDDEN + 255) / 256, 256, 0, stream>>>(W1, W1p, IN_F, HIDDEN);
  pack_w_kernel<<<(HIDDEN * HIDDEN + 255) / 256, 256, 0, stream>>>(W2, W2p, HIDDEN, HIDDEN);
  pack_w_kernel<<<(HIDDEN * OUT_F + 255) / 256, 256, 0, stream>>>(W3, W3p, HIDDEN, OUT_F);
  init_state_kernel<<<(unsigned)((STATE_N + 255) / 256), 256, 0, stream>>>(noisy, x, y, p, z, scalars);

  const unsigned lds1 = 16 * IN_F * 2;        // 72 KB A-slab (GEMM1)
  const unsigned lds2 = 16 * HIDDEN * 2;      // 8 KB A-slab (GEMM2/3)
  for (int n = 0; n < T_STEPS; ++n) {
    float a_n = (float)n / ((float)n + 3.0f);
    float* p_out = (n == T_STEPS - 1) ? out : nullptr;
    step_elem_kernel<<<NB1, 256, 0, stream>>>(x, y, p, z, uv, scalars, inp, a_n,
                                              resP, deltaP, p_out);
    gemm_bf16_kernel<true><<<B_ROWS / 16, 128, lds1, stream>>>(inp, W1p, b1, h1, IN_F, HIDDEN);
    gemm_bf16_kernel<true><<<B_ROWS / 16, 128, lds2, stream>>>(h1, W2p, b2, h2, HIDDEN, HIDDEN);
    gemm3_kernel<<<dim3(B_ROWS / 16, 5), 128, lds2, stream>>>(h2, W3p, b3, uv, qP, HIDDEN, OUT_F);
    finalize_kernel<<<1, 256, 0, stream>>>(resP, NB1, deltaP, NB1, qP, NBQ,
                                           scalars, out + STATE_N + n);
  }
}
